// SwinTransformerBlock_87273735454878
// MI455X (gfx1250) — matrix operand order repaired and checked by reading, not compiled, not run on hardware
//
#include <hip/hip_runtime.h>
#include <math.h>

typedef __bf16 bf16;
typedef __attribute__((ext_vector_type(16))) __bf16 v16bf;
typedef __attribute__((ext_vector_type(8)))  __bf16 v8bf;
typedef __attribute__((ext_vector_type(8)))  float  v8f;

// Problem constants
#define Bd   16
#define Nd   4096
#define Cd   512
#define Hd   8
#define WSd  8
#define SSd  4
#define HDd  64
#define NWd  512          // N/WS
#define TOK  65536        // B*N tokens
#define SCALEv 0.125f     // 64^-0.5

// ---------------------------------------------------------------------------
// Async global -> LDS copy of 16 bytes per lane (CDNA5 async path, ASYNCcnt).
// lds_off is the LDS byte offset (low 32 bits of the flat shared address).
// ---------------------------------------------------------------------------
__device__ __forceinline__ void async_copy_b128(const bf16* g, unsigned lds_off) {
    asm volatile("global_load_async_to_lds_b128 %0, %1, off"
                 :: "v"(lds_off), "v"((unsigned long long)(uintptr_t)g)
                 : "memory");
}

// ---------------------------------------------------------------------------
// Weight convert + transpose: Wt[n][k] = (bf16) W[k][n]
// ---------------------------------------------------------------------------
__global__ __launch_bounds__(256) void wconv_kernel(const float* __restrict__ W,
                                                    bf16* __restrict__ Wt,
                                                    int K, int N) {
    int idx = blockIdx.x * 256 + threadIdx.x;      // 0 .. K*N-1 (exact)
    int n = idx / K;
    int k = idx - n * K;
    Wt[(size_t)n * K + k] = (bf16)W[(size_t)k * N + n];
}

// ---------------------------------------------------------------------------
// LayerNorm (one wave per token, 512 channels).
// permute=1: destination row r is in window order; source is
//            x[b, (n_s + SS) mod N] implementing roll(-SS) + window_partition.
// ---------------------------------------------------------------------------
__global__ __launch_bounds__(256) void ln_kernel(const float* __restrict__ x,
                                                 const float* __restrict__ gw,
                                                 const float* __restrict__ gb,
                                                 bf16* __restrict__ out,
                                                 int permute) {
    int r    = blockIdx.x * 8 + (threadIdx.x >> 5);
    int lane = threadIdx.x & 31;

    int src;
    if (permute) {
        int b2 = r >> 3, t = r & 7;
        int bb = b2 >> 9;            // b2 / NW
        int q  = b2 & 511;           // b2 % NW
        int u  = q * 8 + t;          // = w*NW + nw
        int w  = u >> 9;
        int nw = u & 511;
        int ns = nw * 8 + w;         // token index in shifted sequence
        int n  = (ns + SSd) & (Nd - 1);
        src = bb * Nd + n;
    } else {
        src = r;
    }

    const float* xs = x + (size_t)src * Cd + lane * 16;
    float f[16];
#pragma unroll
    for (int u = 0; u < 4; ++u) {
        float4 v = ((const float4*)xs)[u];
        f[u*4+0] = v.x; f[u*4+1] = v.y; f[u*4+2] = v.z; f[u*4+3] = v.w;
    }
    float s = 0.f;
#pragma unroll
    for (int u = 0; u < 16; ++u) s += f[u];
#pragma unroll
    for (int off = 16; off >= 1; off >>= 1) s += __shfl_xor(s, off, 32);
    float mu = s * (1.0f / Cd);

    float s2 = 0.f;
#pragma unroll
    for (int u = 0; u < 16; ++u) { float d = f[u] - mu; s2 += d * d; }
#pragma unroll
    for (int off = 16; off >= 1; off >>= 1) s2 += __shfl_xor(s2, off, 32);
    float rs = rsqrtf(s2 * (1.0f / Cd) + 1e-5f);

    int c0 = lane * 16;
    bf16* dst = out + (size_t)r * Cd + c0;
#pragma unroll
    for (int half = 0; half < 2; ++half) {
        v8bf o;
#pragma unroll
        for (int u = 0; u < 8; ++u) {
            int c = c0 + half * 8 + u;
            o[u] = (bf16)((f[half*8+u] - mu) * rs * gw[c] + gb[c]);
        }
        *(v8bf*)(dst + half * 8) = o;
    }
}

// ---------------------------------------------------------------------------
// WMMA GEMM: C[M,N] = A[M,K] * Bt[N,K]^T (+ epilogue).
// Block tile 128x128, 4 waves, wave tile 64x64 (16 accumulators), K-step 32.
// Double-buffered LDS, staged with global_load_async_to_lds_b128 (ASYNCcnt).
// MODE 0: QKV   (bias, fold SCALE into Q cols, store bf16 stride 1536)
// MODE 1: proj  (bias + inverse window permute + roll(+SS) + residual -> f32 x1)
// MODE 2: MLP1  (bias + exact GELU, store bf16)
// MODE 3: MLP2  (bias + residual x1, store f32 out)
// ---------------------------------------------------------------------------
template <int MODE, int KDIM, int NDIM>
__global__ __launch_bounds__(128) void gemm_wmma_kernel(
        const bf16*  __restrict__ A,
        const bf16*  __restrict__ Bt,
        const float* __restrict__ bias,
        const float* __restrict__ resid,
        float*       __restrict__ outf,
        bf16*        __restrict__ outb) {
    // 80 B row pitch (bank stagger), two buffers each for A and B tiles.
    __shared__ __align__(16) bf16 As[2][128 * 40];
    __shared__ __align__(16) bf16 Bs[2][128 * 40];

    const int t     = threadIdx.x;
    const int lane  = t & 31;
    const int wid   = t >> 5;
    const int waveM = wid >> 1;
    const int waveN = wid & 1;
    const int l15   = lane & 15;

    const bf16* pA = A  + ((size_t)blockIdx.y * 128 + t) * KDIM;
    const bf16* pB = Bt + ((size_t)blockIdx.x * 128 + t) * KDIM;

    // LDS byte offsets for this thread's staging row (thread t owns row t).
    const unsigned ldsA = (unsigned)(uintptr_t)&As[0][0] + (unsigned)t * 80u;
    const unsigned ldsB = (unsigned)(uintptr_t)&Bs[0][0] + (unsigned)t * 80u;

    auto issue_tile = [&](int k0, int buf) {
#pragma unroll
        for (int u = 0; u < 4; ++u) {
            async_copy_b128(pA + k0 + u * 8, ldsA + (unsigned)buf * 10240u + u * 16u);
            async_copy_b128(pB + k0 + u * 8, ldsB + (unsigned)buf * 10240u + u * 16u);
        }
    };

    const v8f vzero = {0.f,0.f,0.f,0.f,0.f,0.f,0.f,0.f};
    v8f acc[4][4];
#pragma unroll
    for (int i = 0; i < 4; ++i)
#pragma unroll
        for (int j = 0; j < 4; ++j) acc[i][j] = vzero;

    issue_tile(0, 0);                                     // prologue

    for (int k0 = 0; k0 < KDIM; k0 += 32) {
        const int buf = (k0 >> 5) & 1;
        if (k0 + 32 < KDIM) {
            issue_tile(k0 + 32, buf ^ 1);                 // prefetch next tile
            asm volatile("s_wait_asynccnt 0x8" ::: "memory");  // current tile done
        } else {
            asm volatile("s_wait_asynccnt 0x0" ::: "memory");
        }
        __syncthreads();

        // A fragments: 16x32, lane<16 -> K{0..7,16..23}, lane>=16 -> K{8..15,24..31}
        v16bf af[4], bfr[4];
#pragma unroll
        for (int i = 0; i < 4; ++i) {
            int row = waveM * 64 + i * 16 + l15;
            const bf16* rp = &As[buf][row * 40] + ((lane & 16) >> 1);
            v8bf lo = *(const v8bf*)rp;
            v8bf hi = *(const v8bf*)(rp + 16);
            af[i] = __builtin_shufflevector(lo, hi,
                      0,1,2,3,4,5,6,7,8,9,10,11,12,13,14,15);
        }
        // B fragments: 32x16, col = lane%16, lane<16 -> K{0..7,16..23}, lane>=16 -> K{8..15,24..31}
#pragma unroll
        for (int j = 0; j < 4; ++j) {
            int col = waveN * 64 + j * 16 + l15;
            const bf16* rp = &Bs[buf][col * 40] + ((lane & 16) >> 1);
            v8bf lo = *(const v8bf*)rp;
            v8bf hi = *(const v8bf*)(rp + 16);
            bfr[j] = __builtin_shufflevector(lo, hi,
                      0,1,2,3,4,5,6,7,8,9,10,11,12,13,14,15);
        }
#pragma unroll
        for (int i = 0; i < 4; ++i)
#pragma unroll
            for (int j = 0; j < 4; ++j)
                acc[i][j] = __builtin_amdgcn_wmma_f32_16x16x32_bf16(
                    false, af[i], false, bfr[j], (short)0, acc[i][j],
                    false, false);
        __syncthreads();   // before next iteration's async writes reuse buffers
    }

    // Epilogue.  C 16x16 layout: VGPR p, lanes0-15 -> M=p, lanes16-31 -> M=p+8.
    const int rbase = blockIdx.y * 128 + waveM * 64 + ((lane & 16) >> 1);
    const int cbase = blockIdx.x * 128 + waveN * 64 + l15;
#pragma unroll
    for (int i = 0; i < 4; ++i) {
#pragma unroll
        for (int j = 0; j < 4; ++j) {
            int col = cbase + j * 16;
            float bv = bias[col];
#pragma unroll
            for (int p = 0; p < 8; ++p) {
                int   row = rbase + i * 16 + p;
                float val = acc[i][j][p] + bv;
                if (MODE == 0) {                       // QKV
                    if (col < 512) val *= SCALEv;      // fold attn scale into Q
                    outb[(size_t)row * NDIM + col] = (bf16)val;
                } else if (MODE == 1) {                // proj + unpermute + residual
                    int b2 = row >> 3, tt = row & 7;
                    int bb = b2 >> 9, q = b2 & 511;
                    int u  = q * 8 + tt;
                    int w  = u >> 9, nw = u & 511;
                    int n  = ((nw * 8 + w) + SSd) & (Nd - 1);
                    size_t di = ((size_t)bb * Nd + n) * Cd + col;
                    outf[di] = resid[di] + val;
                } else if (MODE == 2) {                // MLP fc1 + exact GELU
                    float g = 0.5f * val * (1.0f + erff(val * 0.70710678118f));
                    outb[(size_t)row * NDIM + col] = (bf16)g;
                } else {                               // MLP fc2 + residual
                    size_t di = (size_t)row * NDIM + col;
                    outf[di] = resid[di] + val;
                }
            }
        }
    }
}

// ---------------------------------------------------------------------------
// Window attention: wave = one (window b2, head h).  8x8 scores over HD=64.
// QKV rows are window-ordered: row = b2*8+t, cols [q|k|v] = m*512 + h*64 + d.
// ---------------------------------------------------------------------------
__device__ __forceinline__ int mw_code(int nw, int idx) {
    // faithful replica of img.reshape(512,8).T.reshape(512,8)
    int u = nw * 8 + idx;
    int n = (u & 511) * 8 + (u >> 9);
    return (n >= Nd - SSd) ? 2 : ((n >= Nd - WSd) ? 1 : 0);
}

__global__ __launch_bounds__(256) void attn_kernel(const bf16* __restrict__ QKV,
                                                   bf16* __restrict__ AO) {
    __shared__ float sm[8 * 1600];                  // per-wave Q,K,V,S
    const int wid  = threadIdx.x >> 5;
    const int lane = threadIdx.x & 31;
    const int b2   = blockIdx.x;
    const int h    = wid;

    float* Q = sm + wid * 1600;
    float* K = Q + 512;
    float* V = K + 512;
    float* S = V + 512;

    const int rr = lane >> 2;                       // token row 0..7
    const int d0 = (lane & 3) * 16;                 // 16 dims per lane

#pragma unroll
    for (int m = 0; m < 3; ++m) {
        const bf16* p = QKV + ((size_t)(b2 * 8 + rr)) * 1536 + m * 512 + h * 64 + d0;
        float* dst = (m == 0 ? Q : (m == 1 ? K : V)) + rr * 64 + d0;
        v8bf lo = ((const v8bf*)p)[0];
        v8bf hi = ((const v8bf*)p)[1];
#pragma unroll
        for (int u = 0; u < 8; ++u) { dst[u] = (float)lo[u]; dst[8+u] = (float)hi[u]; }
    }
    __syncthreads();

    const int nw = b2 & 511;
#pragma unroll
    for (int s = lane; s < 64; s += 32) {
        int i = s >> 3, j = s & 7;
        float a = 0.f;
#pragma unroll
        for (int d = 0; d < 64; ++d) a += Q[i*64+d] * K[j*64+d];   // scale in Q
        if (mw_code(nw, i) != mw_code(nw, j)) a += -100.0f;
        S[s] = a;
    }
    __syncthreads();

    if (lane < 8) {
        int i = lane;
        float mx = -1e30f;
#pragma unroll
        for (int j = 0; j < 8; ++j) mx = fmaxf(mx, S[i*8+j]);
        float e[8], sum = 0.f;
#pragma unroll
        for (int j = 0; j < 8; ++j) { e[j] = expf(S[i*8+j] - mx); sum += e[j]; }
        float inv = 1.0f / sum;
#pragma unroll
        for (int j = 0; j < 8; ++j) S[i*8+j] = e[j] * inv;
    }
    __syncthreads();

    float o[16];
#pragma unroll
    for (int u = 0; u < 16; ++u) o[u] = 0.f;
#pragma unroll
    for (int j = 0; j < 8; ++j) {
        float pj = S[rr*8+j];
#pragma unroll
        for (int u = 0; u < 16; ++u) o[u] += pj * V[j*64 + d0 + u];
    }
    bf16* dst = AO + ((size_t)(b2 * 8 + rr)) * Cd + h * 64 + d0;
    v8bf o0, o1;
#pragma unroll
    for (int u = 0; u < 8; ++u) { o0[u] = (bf16)o[u]; o1[u] = (bf16)o[8+u]; }
    ((v8bf*)dst)[0] = o0;
    ((v8bf*)dst)[1] = o1;
}

// ---------------------------------------------------------------------------
// Host launcher
// ---------------------------------------------------------------------------
extern "C" void kernel_launch(void* const* d_in, const int* in_sizes, int n_in,
                              void* d_out, int out_size, void* d_ws, size_t ws_size,
                              hipStream_t stream) {
    const float* x       = (const float*)d_in[0];
    const float* norm1_w = (const float*)d_in[1];
    const float* norm1_b = (const float*)d_in[2];
    const float* qkv_w   = (const float*)d_in[3];
    const float* qkv_b   = (const float*)d_in[4];
    const float* proj_w  = (const float*)d_in[5];
    const float* proj_b  = (const float*)d_in[6];
    const float* norm2_w = (const float*)d_in[7];
    const float* norm2_b = (const float*)d_in[8];
    const float* mlp_w1  = (const float*)d_in[9];
    const float* mlp_b1  = (const float*)d_in[10];
    const float* mlp_w2  = (const float*)d_in[11];
    const float* mlp_b2  = (const float*)d_in[12];
    float* out = (float*)d_out;

    char* ws = (char*)d_ws;
    size_t off = 0;
    bf16* WqkvT  = (bf16*)(ws + off); off += (size_t)1536 * 512 * 2;
    bf16* WprojT = (bf16*)(ws + off); off += (size_t)512  * 512 * 2;
    bf16* Wm1T   = (bf16*)(ws + off); off += (size_t)2048 * 512 * 2;
    bf16* Wm2T   = (bf16*)(ws + off); off += (size_t)512  * 2048 * 2;
    bf16* Xw     = (bf16*)(ws + off); off += (size_t)TOK * 512 * 2;
    bf16* QKVb   = (bf16*)(ws + off); off += (size_t)TOK * 1536 * 2;
    bf16* AO     = (bf16*)(ws + off); off += (size_t)TOK * 512 * 2;
    float* X1    = (float*)(ws + off); off += (size_t)TOK * 512 * 4;
    bf16* Xn     = (bf16*)(ws + off); off += (size_t)TOK * 512 * 2;
    bf16* H1     = (bf16*)(ws + off); off += (size_t)TOK * 2048 * 2;

    // 1) weight convert + transpose to [N][K] bf16
    wconv_kernel<<<(512*1536)/256, 256, 0, stream>>>(qkv_w,  WqkvT,  512, 1536);
    wconv_kernel<<<(512*512 )/256, 256, 0, stream>>>(proj_w, WprojT, 512, 512);
    wconv_kernel<<<(512*2048)/256, 256, 0, stream>>>(mlp_w1, Wm1T,   512, 2048);
    wconv_kernel<<<(2048*512)/256, 256, 0, stream>>>(mlp_w2, Wm2T,   2048, 512);

    // 2) LN1 + roll(-SS) + window permute -> Xw (window-ordered bf16)
    ln_kernel<<<TOK/8, 256, 0, stream>>>(x, norm1_w, norm1_b, Xw, 1);

    // 3) QKV GEMM (scale folded into Q)
    gemm_wmma_kernel<0, 512, 1536><<<dim3(1536/128, TOK/128), 128, 0, stream>>>(
        Xw, WqkvT, qkv_b, nullptr, nullptr, QKVb);

    // 4) window attention
    attn_kernel<<<TOK/8, 256, 0, stream>>>(QKVb, AO);

    // 5) proj GEMM + inverse permute + roll(+SS) + residual -> X1 (f32)
    gemm_wmma_kernel<1, 512, 512><<<dim3(512/128, TOK/128), 128, 0, stream>>>(
        AO, WprojT, proj_b, x, X1, nullptr);

    // 6) LN2 (natural order) -> Xn
    ln_kernel<<<TOK/8, 256, 0, stream>>>(X1, norm2_w, norm2_b, Xn, 0);

    // 7) MLP fc1 + GELU -> H1
    gemm_wmma_kernel<2, 512, 2048><<<dim3(2048/128, TOK/128), 128, 0, stream>>>(
        Xn, Wm1T, mlp_b1, nullptr, nullptr, H1);

    // 8) MLP fc2 + residual -> out
    gemm_wmma_kernel<3, 2048, 512><<<dim3(512/128, TOK/128), 128, 0, stream>>>(
        H1, Wm2T, mlp_b2, X1, out, nullptr);

    (void)in_sizes; (void)n_in; (void)out_size; (void)ws_size;
}
